// GranularityAttention_54778012893929
// MI455X (gfx1250) — matrix operand order repaired and checked by reading, not compiled, not run on hardware
//
#include <hip/hip_runtime.h>
#include <hip/hip_bf16.h>
#include <math.h>

// ---------------------------------------------------------------------------
// GranularityAttention for MI455X (gfx1250, wave32, WMMA).
//   B=2048, T=512, G=3, H=32, D_MODEL=128.
//
// Kernel 1 (gru_attn_fused): 3 waves per block, one wave per (16-row tile, g).
//   GRU recurrence:  GH = A_whh(16x32 f16) x h^T(32x16 f16) + C
//                    [v_wmma_f32_16x16x32_f16]
//   rank-1 input gates: GI = [Wih|bias](16x4 f32) x [x|1]^T(4x16 f32)
//                    [v_wmma_f32_16x16x4_f32]
//   r/z tiles chain GI directly into the f16 WMMA's C operand.
//   C/D layout: lane -> batch column, VGPR j -> gate row.  h stays f32 in
//   C-layout; next-step B operand built by packing h to f16 first, then a
//   single 4-dword __shfl_xor(16) half-swap (minimal DS traffic on chain).
//   Per-wave VGPR budget ~220 (< 256: no s_set_vgpr_msb windowing).
//   Epilogue: scores/softmax/weighted across the 3 g-waves via LDS.
//
// Kernel 2 (broadcast_out): out = weighted @ Wo + bo, replicated over T with
//   non-temporal b128 stores (512 MiB stream => ~23us floor @ 23.3 TB/s).
// ---------------------------------------------------------------------------

typedef __attribute__((ext_vector_type(16))) _Float16 v16h;
typedef __attribute__((ext_vector_type(2)))  __fp16   v2fp16;  // cvt_pkrtz result type
typedef __attribute__((ext_vector_type(8)))  float    v8f;
typedef __attribute__((ext_vector_type(2)))  float    v2f;
typedef __attribute__((ext_vector_type(4)))  float    f4;
typedef __attribute__((ext_vector_type(8)))  int      v8i;

#define GG 3
#define HH 32
#define DM 128

#if __has_builtin(__builtin_amdgcn_rcpf)
__device__ __forceinline__ float fast_rcp(float x) { return __builtin_amdgcn_rcpf(x); }
#else
__device__ __forceinline__ float fast_rcp(float x) { return 1.0f / x; }
#endif

#if __has_builtin(__builtin_amdgcn_tanhf)
__device__ __forceinline__ float fast_tanh(float x) { return __builtin_amdgcn_tanhf(x); }
// sigma(x) = 0.5*tanh(x/2) + 0.5 : one TRANS op instead of exp+rcp.
__device__ __forceinline__ float sigmoid_(float x) {
    return __builtin_fmaf(0.5f, __builtin_amdgcn_tanhf(0.5f * x), 0.5f);
}
#else
__device__ __forceinline__ float fast_tanh(float x) { return tanhf(x); }
__device__ __forceinline__ float sigmoid_(float x) {
    return fast_rcp(1.0f + __expf(-x));
}
#endif

__global__ __launch_bounds__(96) void gru_attn_fused(
    const float* __restrict__ x_mom,   // (B,T,3)
    const float* __restrict__ context, // (B,128)
    const float* __restrict__ W_ih,    // (3,96)
    const float* __restrict__ W_hh,    // (3,96,32)
    const float* __restrict__ b_ih,    // (3,96)
    const float* __restrict__ b_hh,    // (3,96)
    const float* __restrict__ Wq,      // (128,32)
    const float* __restrict__ bq,      // (32)
    const float* __restrict__ Ws,      // (32)
    const float* __restrict__ bs,      // (1)
    const float* __restrict__ logT,    // (1)
    float* __restrict__ weighted_ws,   // (B,32) scratch
    float* __restrict__ attn_out,      // (B,3)  -> tail of d_out
    int B, int T)
{
    const int lane = threadIdx.x & 31;  // 0..31 (wave32)
    const int g    = threadIdx.x >> 5;  // wave id == granularity index
    const int bc   = lane & 15;         // batch column within tile (N index)
    const int half = lane >> 4;         // 0: lanes 0-15, 1: lanes 16-31
    const int off  = half * 8;          // gate-row offset carried by this lane half
    const int b    = blockIdx.x * 16 + bc;

    __shared__ float hsh[GG][HH][16];   // h_final[g][k][bc] for the epilogue
    __shared__ float ssc[GG][16];       // scores[g][bc]

    const v8f vzero = {};

    // ---- loop-invariant WMMA A operands (register resident, this g only) ----
    // A_whh[nt] : 16x32 f16 tile, rows = gates nt*16..nt*16+15, cols = hidden k.
    // f16 A layout: lane(l&15)=row M; element e -> K = e + 16*half.
    v16h Awhh[6];
    // A_gi[nt] : 16x4 f32 tile. half 0 holds K0,K1 = (Wih, bias); half 1: zeros.
    // bias = b_ih + b_hh for r,z tiles (nt<4), b_ih for n tiles (b_hh stays in GH).
    v2f Agi[6];
    // b_hh seed for the n-gate GH tiles. C layout: VGPR j -> gate row 16*nt+j+off.
    v8f Cbhh[2];

#pragma unroll
    for (int nt = 0; nt < 6; ++nt) {
        const int gate = nt * 16 + bc;
        const float* wrow = W_hh + ((size_t)g * 96 + gate) * HH;
#pragma unroll
        for (int e = 0; e < 16; ++e) {
            const int k = e + 16 * half;
            Awhh[nt][e] = (_Float16)wrow[k];
        }
        const float wih = W_ih[g * 96 + gate];
        const float bih = b_ih[g * 96 + gate];
        const float bhh = b_hh[g * 96 + gate];
        v2f a;
        a[0] = half ? 0.0f : wih;
        a[1] = half ? 0.0f : (nt < 4 ? (bih + bhh) : bih);
        Agi[nt] = a;
    }
#pragma unroll
    for (int p = 0; p < 2; ++p) {
        v8f c = vzero;
#pragma unroll
        for (int j = 0; j < 8; ++j)
            c[j] = b_hh[g * 96 + (4 + p) * 16 + j + off];
        Cbhh[p] = c;
    }

    // h state, C layout: h[p][j] = h[hidden k = 16*p + j + off][batch bc], f32.
    float h[2][8] = {};

    const float* xrow = x_mom + (size_t)b * T * GG + g;

    for (int t = 0; t < T; ++t) {
        __builtin_prefetch(xrow + 3 * (t + 16), 0, 1);   // global_prefetch hint
        const float xv = xrow[3 * t];

        // ---- build B operand h^T (32x16 f16): lane -> column N=bc,
        //      element e -> K = e + 16*half.  Lane l owns k in {0-7,16-23},
        //      lane l+16 owns {8-15,24-31}.  Pack to f16 FIRST, then one
        //      4-dword half-swap via __shfl_xor(16).
        int P0[4], P1[4];
#pragma unroll
        for (int i = 0; i < 4; ++i) {
            const v2fp16 p0 = __builtin_amdgcn_cvt_pkrtz(h[0][2 * i], h[0][2 * i + 1]);
            const v2fp16 p1 = __builtin_amdgcn_cvt_pkrtz(h[1][2 * i], h[1][2 * i + 1]);
            P0[i] = __builtin_bit_cast(int, p0);
            P1[i] = __builtin_bit_cast(int, p1);
        }
        v8i bh32;
#pragma unroll
        for (int i = 0; i < 4; ++i) {
            const int send = half ? P0[i] : P1[i];   // give away the other half's K range
            const int recv = __shfl_xor(send, 16, 32);
            bh32[i]     = half ? recv  : P0[i];      // K = 2i,2i+1   (+16*half)
            bh32[i + 4] = half ? P1[i] : recv;       // K = 8+2i,9+2i (+16*half)
        }
        const v16h Bh = __builtin_bit_cast(v16h, bh32);

        // rank-1 gi operand: B = [x | 1 | 0 | 0]^T (4x16 f32)
        v2f Bgi;
        Bgi[0] = half ? 0.0f : xv;
        Bgi[1] = half ? 0.0f : 1.0f;

        // GI = x*Wih + bias (per gate row, broadcast over batch columns)
        v8f GI[6];
#pragma unroll
        for (int nt = 0; nt < 6; ++nt)
            GI[nt] = __builtin_amdgcn_wmma_f32_16x16x4_f32(
                false, Agi[nt], false, Bgi, (short)0, vzero, false, false);

        // GH: r/z tiles chain GI through the C operand; n tiles seed with b_hh.
        v8f GH[6];
#pragma unroll
        for (int nt = 0; nt < 6; ++nt) {
            const v8f c0 = (nt < 4) ? GI[nt] : Cbhh[nt - 4];
            GH[nt] = __builtin_amdgcn_wmma_f32_16x16x32_f16(
                false, Awhh[nt], false, Bh, (short)0, c0, false, false);
        }

        // ---- GRU gate math (f32). Tile nt holds gates 16*nt + j + off.
        // hidden k = 16*p + j + off:  r @ tile p, z @ tile 2+p, n @ tile 4+p.
#pragma unroll
        for (int p = 0; p < 2; ++p)
#pragma unroll
            for (int j = 0; j < 8; ++j) {
                const float r = sigmoid_(GH[p][j]);          // includes GI
                const float z = sigmoid_(GH[2 + p][j]);      // includes GI
                const float n = fast_tanh(GI[4 + p][j] + r * GH[4 + p][j]);
                h[p][j] = (1.0f - z) * n + z * h[p][j];
            }
    }

    // ------------------- attention epilogue (once) -------------------
    // query[k] = context[b] . Wq[:,k] + bq[k]   for this lane's 16 k values.
    float q[2][8];
#pragma unroll
    for (int p = 0; p < 2; ++p)
#pragma unroll
        for (int j = 0; j < 8; ++j)
            q[p][j] = bq[16 * p + j + off];
    {
        const float* ctx = context + (size_t)b * DM;
        for (int d = 0; d < DM; ++d) {
            const float cd = ctx[d];
#pragma unroll
            for (int p = 0; p < 2; ++p)
#pragma unroll
                for (int j = 0; j < 8; ++j)
                    q[p][j] += cd * Wq[d * HH + 16 * p + j + off];
        }
    }

    // score for this g; also publish h to LDS for the cross-g reduction.
    float part = 0.0f;
#pragma unroll
    for (int p = 0; p < 2; ++p)
#pragma unroll
        for (int j = 0; j < 8; ++j) {
            const int k = 16 * p + j + off;
            hsh[g][k][bc] = h[p][j];
            part += fast_tanh(h[p][j] + q[p][j]) * Ws[k];
        }
    part += __shfl_xor(part, 16, 32);   // combine the two lane halves
    if (half == 0) ssc[g][bc] = part + bs[0];

    __syncthreads();

    if (g == 0) {
        const float temp = fmaxf(__expf(logT[0]), 0.1f);
        const float rt   = fast_rcp(temp);
        const float s0 = ssc[0][bc] * rt;
        const float s1 = ssc[1][bc] * rt;
        const float s2 = ssc[2][bc] * rt;
        const float m  = fmaxf(s0, fmaxf(s1, s2));
        const float e0 = __expf(s0 - m), e1 = __expf(s1 - m), e2 = __expf(s2 - m);
        const float inv = fast_rcp(e0 + e1 + e2);
        const float a0 = e0 * inv, a1 = e1 * inv, a2 = e2 * inv;

        if (half == 0) {
            attn_out[(size_t)b * 3 + 0] = a0;
            attn_out[(size_t)b * 3 + 1] = a1;
            attn_out[(size_t)b * 3 + 2] = a2;
        }

#pragma unroll
        for (int p = 0; p < 2; ++p)
#pragma unroll
            for (int j = 0; j < 8; ++j) {
                const int k = 16 * p + j + off;
                const float wv = hsh[0][k][bc] * a0
                               + hsh[1][k][bc] * a1
                               + hsh[2][k][bc] * a2;
                weighted_ws[(size_t)b * HH + k] = wv;
            }
    }
}

// ---------------------------------------------------------------------------
// Kernel 2: out[b,t,:] = weighted[b] @ Wo + bo, replicated over t.
// One block per batch row; 128 threads; streaming non-temporal b128 stores.
// ---------------------------------------------------------------------------
__global__ __launch_bounds__(128) void broadcast_out(
    const float* __restrict__ weighted, // (B,32)
    const float* __restrict__ Wo,       // (32,128)
    const float* __restrict__ bo,       // (128)
    float* __restrict__ out,            // (B,T,128)
    int T)
{
    const int b   = blockIdx.x;
    const int tid = threadIdx.x;

    __shared__ float wsh[HH];
    __shared__ float row[DM];

    if (tid < HH) wsh[tid] = weighted[(size_t)b * HH + tid];
    __syncthreads();

    {
        float acc = bo[tid];
#pragma unroll
        for (int k = 0; k < HH; ++k)
            acc += wsh[k] * Wo[k * DM + tid];
        row[tid] = acc;
    }
    __syncthreads();

    const int c4 = (tid & 31) * 4;   // float4 column within the 128-float row
    const int tr = tid >> 5;         // 0..3 starting t offset
    f4 v4;
    v4[0] = row[c4 + 0]; v4[1] = row[c4 + 1];
    v4[2] = row[c4 + 2]; v4[3] = row[c4 + 3];

    float* base = out + (size_t)b * T * DM;
    for (int t = tr; t < T; t += 4) {
        __builtin_nontemporal_store(v4, (f4*)(base + (size_t)t * DM + c4));
    }
}

// ---------------------------------------------------------------------------
extern "C" void kernel_launch(void* const* d_in, const int* in_sizes, int n_in,
                              void* d_out, int out_size, void* d_ws, size_t ws_size,
                              hipStream_t stream) {
    const float* x_mom   = (const float*)d_in[0];
    const float* context = (const float*)d_in[1];
    const float* W_ih    = (const float*)d_in[2];
    const float* W_hh    = (const float*)d_in[3];
    const float* b_ih    = (const float*)d_in[4];
    const float* b_hh    = (const float*)d_in[5];
    const float* Wq      = (const float*)d_in[6];
    const float* bq      = (const float*)d_in[7];
    const float* Ws      = (const float*)d_in[8];
    const float* bs      = (const float*)d_in[9];
    const float* Wo      = (const float*)d_in[10];
    const float* bo      = (const float*)d_in[11];
    const float* logT    = (const float*)d_in[12];

    const int B = in_sizes[1] / DM;            // context is (B,128)
    const int T = in_sizes[0] / (B * GG);      // x_mom is (B,T,3)

    float* out      = (float*)d_out;
    float* attn_out = out + (size_t)B * T * DM;   // tuple tail: (B,3)
    float* weighted = (float*)d_ws;               // (B,32) scratch

    gru_attn_fused<<<B / 16, 96, 0, stream>>>(
        x_mom, context, W_ih, W_hh, b_ih, b_hh, Wq, bq, Ws, bs, logT,
        weighted, attn_out, B, T);

    broadcast_out<<<B, 128, 0, stream>>>(weighted, Wo, bo, out, T);
}
